// Model_84868553769102
// MI455X (gfx1250) — compile-verified, matrix operand order checked by reading, not run on hardware
//
#include <hip/hip_runtime.h>
#include <math.h>

// ---------------- problem constants ----------------
#define Bsz   32
#define Ssz   512
#define Csz   862
#define Cpad  864
#define MT    54          // 864 / 16 M-tiles (waves process 2 at a time)
#define Dm    128
#define DIx   256         // d_inner
#define Nst   16
#define Ppred 96
#define DTRk  8
#define BCn   (Bsz*Csz)

typedef __attribute__((ext_vector_type(16))) __bf16 v16bf;
typedef __attribute__((ext_vector_type(8)))  float  v8f;

__device__ __forceinline__ unsigned short f32_to_bf16(float f) {
  unsigned int u = __float_as_uint(f);
  u += 0x7FFFu + ((u >> 16) & 1u);          // round-to-nearest-even
  return (unsigned short)(u >> 16);
}

__device__ __forceinline__ v16bf ldfrag(const unsigned short* p) {
  return *(const v16bf*)(const void*)p;
}

// K index inside a 16x32 bf16 A/B fragment for (lane, element j)
__device__ __forceinline__ int frag_k(int lane, int j) {
  return ((j >> 3) << 4) + ((lane >> 4) << 3) + (j & 7);
}

// ---------------- RevIN statistics ----------------
// alpha = w/std, beta = b - w*mean/std ; keep mean/std for denorm
__global__ void revin_stats(const float* __restrict__ x,
                            const float* __restrict__ rw, const float* __restrict__ rb,
                            float* __restrict__ alpha, float* __restrict__ beta,
                            float* __restrict__ meanv, float* __restrict__ stdv) {
  int id = blockIdx.x * 256 + threadIdx.x;
  if (id >= BCn) return;
  int b = id / Csz, c = id % Csz;
  const float* p = x + (size_t)b * Ssz * Csz + c;
  float s = 0.f, ss = 0.f;
  for (int t = 0; t < Ssz; ++t) { float v = p[(size_t)t * Csz]; s += v; ss += v * v; }
  float mean = s * (1.f / Ssz);
  float var  = ss * (1.f / Ssz) - mean * mean;
  float sd   = sqrtf(var + 1e-5f);
  alpha[id] = rw[c] / sd;
  beta[id]  = rb[c] - rw[c] * mean / sd;
  meanv[id] = mean;
  stdv[id]  = sd;
}

// sw1[d] = sum_s w1[d,s]  (for folding RevIN shift into GEMM1)
__global__ void sum_w1(const float* __restrict__ w1, float* __restrict__ sw1) {
  int d = threadIdx.x;
  if (d >= Dm) return;
  float s = 0.f;
  for (int t = 0; t < Ssz; ++t) s += w1[(size_t)d * Ssz + t];
  sw1[d] = s;
}

// ---------------- fragment packing ----------------
// A-fragments from x (transposed source):  A[m=c, k=s] = x[b,s,c]
__global__ void pack_x(const float* __restrict__ x, unsigned short* __restrict__ dst) {
  int b = blockIdx.y;
  size_t id = (size_t)blockIdx.x * 256 + threadIdx.x;   // < 54*16*512
  int j = id & 15, lane = (id >> 4) & 31;
  size_t t = id >> 9;
  int k = t & 15, ct = t >> 4;
  int c = ct * 16 + (lane & 15);
  int s = k * 32 + frag_k(lane, j);
  float v = (c < Csz) ? x[((size_t)b * Ssz + s) * Csz + c] : 0.f;
  dst[(size_t)b * (MT * 16 * 512) + id] = f32_to_bf16(v);
}

// A-fragments from row-major f32 [rows x ld] per batch
__global__ void pack_a(const float* __restrict__ src, unsigned short* __restrict__ dst,
                       int KS, int Mvalid, int Kvalid, int ld, size_t batchStride) {
  int b = blockIdx.y;
  size_t id = (size_t)blockIdx.x * 256 + threadIdx.x;   // < MT*KS*512
  int j = id & 15, lane = (id >> 4) & 31;
  size_t t = id >> 9;
  int k = (int)(t % KS), mt = (int)(t / KS);
  int m  = mt * 16 + (lane & 15);
  int kk = k * 32 + frag_k(lane, j);
  float v = 0.f;
  if (m < Mvalid && kk < Kvalid)
    v = src[(size_t)b * batchStride + (size_t)m * ld + kk];
  dst[(size_t)b * ((size_t)MT * KS * 512) + id] = f32_to_bf16(v);
}

// B-fragments from weight W row-major [N x K]  (GEMM computes A @ W^T)
__global__ void pack_b(const float* __restrict__ W, unsigned short* __restrict__ dst,
                       int KS, int Nvalid, int Kvalid) {
  size_t id = (size_t)blockIdx.x * 256 + threadIdx.x;   // < NT*KS*512
  int j = id & 15, lane = (id >> 4) & 31;
  size_t t = id >> 9;
  int k = (int)(t % KS), nt = (int)(t / KS);
  int n  = nt * 16 + (lane & 15);
  int kk = k * 32 + frag_k(lane, j);
  float v = (n < Nvalid && kk < Kvalid) ? W[(size_t)n * Kvalid + kk] : 0.f;
  dst[id] = f32_to_bf16(v);
}

// -------- WMMA GEMM: one wave = 2 M-tiles x NPW N-tiles, ping-pong k ----
// k unrolled by 2 with two NAMED register sets (no rotation movs): load k+1
// into set B, WMMA burst on set A, load k+2 into set A, WMMA burst on set B.
// Loads are always one burst ahead of their consumers -> partial loadcnt
// waits overlap vmem with matrix math; 6 loads feed 8 WMMAs (NPW=4).
// MODE 0: plain store   MODE 1: RevIN-affine (alpha*v + beta*sw1[n] + b1[n])
// MODE 2: + residual e0[same idx]           MODE 3: + bias e0[n]
template <int MODE, int NPW>
__global__ __launch_bounds__(32)
void gemm_wmma(const unsigned short* __restrict__ Ap, const unsigned short* __restrict__ Bp,
               float* __restrict__ D, int KS, int ldd,
               const float* __restrict__ e0, const float* __restrict__ e1,
               const float* __restrict__ e2, const float* __restrict__ e3) {
  const int lane = threadIdx.x;
  const int mg = blockIdx.x, ng = blockIdx.y, b = blockIdx.z;
  const size_t a0base = (((size_t)b * MT + 2 * mg) * (size_t)KS) * 512 + (size_t)lane * 16;
  const size_t a1base = a0base + (size_t)KS * 512;
  const size_t bbase  = ((size_t)ng * NPW * (size_t)KS) * 512 + (size_t)lane * 16;

  v8f acc[2][NPW];
#pragma unroll
  for (int mi = 0; mi < 2; ++mi)
#pragma unroll
    for (int t = 0; t < NPW; ++t) acc[mi][t] = (v8f){};

  // set A preloaded with k = 0
  v16bf a0A = ldfrag(Ap + a0base);
  v16bf a1A = ldfrag(Ap + a1base);
  v16bf bA[NPW];
#pragma unroll
  for (int t = 0; t < NPW; ++t) bA[t] = ldfrag(Bp + bbase + (size_t)t * KS * 512);

  for (int k = 0; k < KS; k += 2) {            // KS is even for all call sites
    // fetch k+1 into set B
    v16bf a0B = ldfrag(Ap + a0base + (size_t)(k + 1) * 512);
    v16bf a1B = ldfrag(Ap + a1base + (size_t)(k + 1) * 512);
    v16bf bB[NPW];
#pragma unroll
    for (int t = 0; t < NPW; ++t)
      bB[t] = ldfrag(Bp + bbase + ((size_t)t * KS + k + 1) * 512);

    // compute k with set A
#pragma unroll
    for (int t = 0; t < NPW; ++t)
      acc[0][t] = __builtin_amdgcn_wmma_f32_16x16x32_bf16(false, a0A, false, bA[t],
                                                          (short)0, acc[0][t], false, false);
#pragma unroll
    for (int t = 0; t < NPW; ++t)
      acc[1][t] = __builtin_amdgcn_wmma_f32_16x16x32_bf16(false, a1A, false, bA[t],
                                                          (short)0, acc[1][t], false, false);

    // fetch k+2 into set A
    if (k + 2 < KS) {
      a0A = ldfrag(Ap + a0base + (size_t)(k + 2) * 512);
      a1A = ldfrag(Ap + a1base + (size_t)(k + 2) * 512);
#pragma unroll
      for (int t = 0; t < NPW; ++t)
        bA[t] = ldfrag(Bp + bbase + ((size_t)t * KS + k + 2) * 512);
    }

    // compute k+1 with set B
#pragma unroll
    for (int t = 0; t < NPW; ++t)
      acc[0][t] = __builtin_amdgcn_wmma_f32_16x16x32_bf16(false, a0B, false, bB[t],
                                                          (short)0, acc[0][t], false, false);
#pragma unroll
    for (int t = 0; t < NPW; ++t)
      acc[1][t] = __builtin_amdgcn_wmma_f32_16x16x32_bf16(false, a1B, false, bB[t],
                                                          (short)0, acc[1][t], false, false);
  }

  const int mhi = (lane >> 4) << 3;
#pragma unroll
  for (int mi = 0; mi < 2; ++mi) {
#pragma unroll
    for (int t = 0; t < NPW; ++t) {
      const int n = (ng * NPW + t) * 16 + (lane & 15);
#pragma unroll
      for (int r = 0; r < 8; ++r) {
        int row = (2 * mg + mi) * 16 + r + mhi;
        float v = acc[mi][t][r];
        size_t oidx = ((size_t)b * Cpad + row) * (size_t)ldd + n;
        if (MODE == 0) {
          D[oidx] = v;
        } else if (MODE == 1) {
          float o = 0.f;
          if (row < Csz) {
            float al = e0[(size_t)b * Csz + row];
            float be = e1[(size_t)b * Csz + row];
            o = al * v + be * e2[n] + e3[n];
          }
          D[oidx] = o;
        } else if (MODE == 2) {
          D[oidx] = v + e0[oidx];
        } else {
          D[oidx] = v + e0[n];
        }
      }
    }
  }
}

// ---------------- causal depthwise conv (K=4) + SiLU ----------------
__global__ void conv_silu(const float* __restrict__ xz, const float* __restrict__ cw,
                          const float* __restrict__ cb, float* __restrict__ xmc) {
  size_t id = (size_t)blockIdx.x * 256 + threadIdx.x;    // < B*Cpad*DI
  if (id >= (size_t)Bsz * Cpad * DIx) return;
  int d = id % DIx;
  size_t t = id / DIx;
  int l = (int)(t % Cpad), b = (int)(t / Cpad);
  float s = cb[d];
  for (int k = 0; k < 4; ++k) {
    int lp = l - 3 + k;
    if (lp >= 0) s += xz[((size_t)b * Cpad + lp) * 512 + d] * cw[d * 4 + k];
  }
  float sig = 1.f / (1.f + __expf(-s));
  xmc[id] = s * sig;
}

// ---------------- fused selective scan ----------------
// block = one batch, thread = one channel d; N=16 states in registers.
// Per step: dt = softplus(dbl[0:8]·dtW + dtb); dA = exp(dt*A);
// h = dA*h + (dt*x)*B ; y = h·C ; ys = (y + D*x) * silu(z)
__global__ __launch_bounds__(256)
void scan_kernel(const float* __restrict__ dbl, const float* __restrict__ xmc,
                 const float* __restrict__ xz,  const float* __restrict__ dtw,
                 const float* __restrict__ dtb, const float* __restrict__ A_log,
                 const float* __restrict__ Dskip, float* __restrict__ ys) {
  const int b = blockIdx.x, d = threadIdx.x;
  __shared__ float sB[40];
  float dw[8];
  for (int r = 0; r < 8; ++r) dw[r] = dtw[d * DTRk + r];
  const float dtbd = dtb[d];
  float Ar[Nst];
  for (int n = 0; n < Nst; ++n) Ar[n] = -__expf(A_log[d * Nst + n]);
  const float Dv = Dskip[d];
  float st[Nst];
  for (int n = 0; n < Nst; ++n) st[n] = 0.f;

  for (int l = 0; l < Csz; ++l) {
    if (d < 40) sB[d] = dbl[((size_t)b * Cpad + l) * 48 + d];
    __syncthreads();
    if (l + 1 < Csz) {
      __builtin_prefetch(&dbl[((size_t)b * Cpad + l + 1) * 48], 0, 0);
      __builtin_prefetch(&xmc[((size_t)b * Cpad + l + 1) * DIx + d], 0, 0);
    }
    float t = dtbd;
    for (int r = 0; r < 8; ++r) t += sB[r] * dw[r];
    float dt = (t > 20.f) ? t : log1pf(__expf(t));        // softplus
    float xv = xmc[((size_t)b * Cpad + l) * DIx + d];
    float zv = xz[((size_t)b * Cpad + l) * 512 + DIx + d];
    float dtx = dt * xv;
    float y = 0.f;
    for (int n = 0; n < Nst; ++n) {
      float dA = __expf(dt * Ar[n]);
      st[n] = dA * st[n] + dtx * sB[8 + n];
      y += st[n] * sB[24 + n];
    }
    float sig = 1.f / (1.f + __expf(-zv));
    ys[((size_t)b * Cpad + l) * DIx + d] = (y + Dv * xv) * (zv * sig);
    __syncthreads();
  }
}

// ---------------- head bias done in GEMM5; final transpose + RevIN denorm ----
__global__ void final_denorm(const float* __restrict__ o5,
                             const float* __restrict__ rw, const float* __restrict__ rb,
                             const float* __restrict__ meanv, const float* __restrict__ stdv,
                             float* __restrict__ out) {
  size_t id = (size_t)blockIdx.x * 256 + threadIdx.x;    // < B*P*C
  if (id >= (size_t)Bsz * Ppred * Csz) return;
  int c = id % Csz;
  size_t t = id / Csz;
  int p = (int)(t % Ppred), b = (int)(t / Ppred);
  float v = o5[((size_t)b * Cpad + c) * Ppred + p];
  size_t bc = (size_t)b * Csz + c;
  out[id] = (v - rb[c]) / (rw[c] + 1e-10f) * stdv[bc] + meanv[bc];
}

// ---------------- launch ----------------
extern "C" void kernel_launch(void* const* d_in, const int* in_sizes, int n_in,
                              void* d_out, int out_size, void* d_ws, size_t ws_size,
                              hipStream_t stream) {
  const float* x         = (const float*)d_in[0];
  const float* revin_w   = (const float*)d_in[1];
  const float* revin_b   = (const float*)d_in[2];
  const float* w1        = (const float*)d_in[3];
  const float* b1        = (const float*)d_in[4];
  const float* in_proj_w = (const float*)d_in[5];
  const float* conv_w    = (const float*)d_in[6];
  const float* conv_b    = (const float*)d_in[7];
  const float* x_proj_w  = (const float*)d_in[8];
  const float* dt_proj_w = (const float*)d_in[9];
  const float* dt_proj_b = (const float*)d_in[10];
  const float* A_log     = (const float*)d_in[11];
  const float* D_skip    = (const float*)d_in[12];
  const float* out_proj_w= (const float*)d_in[13];
  const float* head_w    = (const float*)d_in[14];
  const float* head_b    = (const float*)d_in[15];
  float* out = (float*)d_out;

  char* ws = (char*)d_ws;
  auto alloc = [&](size_t bytes) -> char* {
    char* p = ws;
    ws += (bytes + 255) & ~(size_t)255;
    return p;
  };

  float* alpha = (float*)alloc((size_t)BCn * 4);
  float* beta  = (float*)alloc((size_t)BCn * 4);
  float* meanv = (float*)alloc((size_t)BCn * 4);
  float* stdv  = (float*)alloc((size_t)BCn * 4);
  float* sw1   = (float*)alloc(Dm * 4);

  unsigned short* BpW1 = (unsigned short*)alloc((size_t)8  * 16 * 512 * 2);
  unsigned short* BpIP = (unsigned short*)alloc((size_t)32 * 4  * 512 * 2);
  unsigned short* BpXP = (unsigned short*)alloc((size_t)3  * 8  * 512 * 2);
  unsigned short* BpOP = (unsigned short*)alloc((size_t)8  * 8  * 512 * 2);
  unsigned short* BpHD = (unsigned short*)alloc((size_t)6  * 4  * 512 * 2);

  unsigned short* Ap1 = (unsigned short*)alloc((size_t)Bsz * MT * 16 * 512 * 2);
  float*          h   = (float*)alloc((size_t)Bsz * Cpad * Dm * 4);
  unsigned short* Ap2 = (unsigned short*)alloc((size_t)Bsz * MT * 4 * 512 * 2);
  float*          xz  = (float*)alloc((size_t)Bsz * Cpad * 512 * 4);
  float*          xmc = (float*)alloc((size_t)Bsz * Cpad * DIx * 4);
  unsigned short* Ap3 = (unsigned short*)alloc((size_t)Bsz * MT * 8 * 512 * 2);
  float*          dbl = (float*)alloc((size_t)Bsz * Cpad * 48 * 4);
  float*          ys  = (float*)alloc((size_t)Bsz * Cpad * DIx * 4);
  unsigned short* Ap4 = (unsigned short*)alloc((size_t)Bsz * MT * 8 * 512 * 2);
  float*          o4  = (float*)alloc((size_t)Bsz * Cpad * Dm * 4);
  unsigned short* Ap5 = (unsigned short*)alloc((size_t)Bsz * MT * 4 * 512 * 2);
  float*          o5  = (float*)alloc((size_t)Bsz * Cpad * Ppred * 4);
  (void)ws_size; (void)in_sizes; (void)n_in; (void)out_size;

  // 1. RevIN stats
  revin_stats<<<(BCn + 255) / 256, 256, 0, stream>>>(x, revin_w, revin_b,
                                                     alpha, beta, meanv, stdv);
  // 2. row sums of w1
  sum_w1<<<1, 128, 0, stream>>>(w1, sw1);

  // 3. pack all weights into B-fragments
  pack_b<<<8  * 16 * 2, 256, 0, stream>>>(w1,        BpW1, 16, 128, 512);
  pack_b<<<32 * 4  * 2, 256, 0, stream>>>(in_proj_w, BpIP, 4,  512, 128);
  pack_b<<<3  * 8  * 2, 256, 0, stream>>>(x_proj_w,  BpXP, 8,  40,  256);
  pack_b<<<8  * 8  * 2, 256, 0, stream>>>(out_proj_w,BpOP, 8,  128, 256);
  pack_b<<<6  * 4  * 2, 256, 0, stream>>>(head_w,    BpHD, 4,  96,  128);

  // 4. pack x (transposed gather) -> A-fragments for GEMM1
  pack_x<<<dim3(MT * 16 * 2, Bsz), 256, 0, stream>>>(x, Ap1);

  // 5. GEMM1: h = RevIN-affine( x^T @ w1^T )      [B,864,128]  2x4 tiles/wave
  gemm_wmma<1, 4><<<dim3(MT / 2, 2, Bsz), 32, 0, stream>>>(Ap1, BpW1, h, 16, Dm,
                                                           alpha, beta, sw1, b1);
  // 6. pack h -> A2
  pack_a<<<dim3(MT * 4 * 2, Bsz), 256, 0, stream>>>(h, Ap2, 4, Cpad, Dm, Dm,
                                                    (size_t)Cpad * Dm);
  // 7. GEMM2: xz = h @ in_proj_w^T                [B,864,512]  2x4 tiles/wave
  gemm_wmma<0, 4><<<dim3(MT / 2, 8, Bsz), 32, 0, stream>>>(Ap2, BpIP, xz, 4, 512,
                                                           nullptr, nullptr, nullptr, nullptr);
  // 8. depthwise causal conv + SiLU -> xmc        [B,864,256]
  {
    size_t tot = (size_t)Bsz * Cpad * DIx;
    conv_silu<<<(unsigned)((tot + 255) / 256), 256, 0, stream>>>(xz, conv_w, conv_b, xmc);
  }
  // 9. pack xmc -> A3
  pack_a<<<dim3(MT * 8 * 2, Bsz), 256, 0, stream>>>(xmc, Ap3, 8, Csz, DIx, DIx,
                                                    (size_t)Cpad * DIx);
  // 10. GEMM3: dbl = xmc @ x_proj_w^T             [B,864,48]   2x3 tiles/wave
  gemm_wmma<0, 3><<<dim3(MT / 2, 1, Bsz), 32, 0, stream>>>(Ap3, BpXP, dbl, 8, 48,
                                                           nullptr, nullptr, nullptr, nullptr);
  // 11. fused selective scan + D-skip + z-gate -> ys [B,864,256]
  scan_kernel<<<Bsz, 256, 0, stream>>>(dbl, xmc, xz, dt_proj_w, dt_proj_b,
                                       A_log, D_skip, ys);
  // 12. pack ys -> A4
  pack_a<<<dim3(MT * 8 * 2, Bsz), 256, 0, stream>>>(ys, Ap4, 8, Csz, DIx, DIx,
                                                    (size_t)Cpad * DIx);
  // 13. GEMM4: o4 = ys @ out_proj_w^T + h (residual)  [B,864,128]  2x4 tiles/wave
  gemm_wmma<2, 4><<<dim3(MT / 2, 2, Bsz), 32, 0, stream>>>(Ap4, BpOP, o4, 8, Dm,
                                                           h, nullptr, nullptr, nullptr);
  // 14. pack o4 -> A5
  pack_a<<<dim3(MT * 4 * 2, Bsz), 256, 0, stream>>>(o4, Ap5, 4, Cpad, Dm, Dm,
                                                    (size_t)Cpad * Dm);
  // 15. GEMM5: o5 = o4 @ head_w^T + head_b        [B,864,96]   2x3 tiles/wave
  gemm_wmma<3, 3><<<dim3(MT / 2, 2, Bsz), 32, 0, stream>>>(Ap5, BpHD, o5, 4, Ppred,
                                                           head_b, nullptr, nullptr, nullptr);
  // 16. transpose [B,C,P]->[B,P,C] + RevIN denorm
  {
    size_t tot = (size_t)Bsz * Ppred * Csz;
    final_denorm<<<(unsigned)((tot + 255) / 256), 256, 0, stream>>>(o5, revin_w, revin_b,
                                                                    meanv, stdv, out);
  }
}
